// MultiHeadCrossAttention_72971494359245
// MI455X (gfx1250) — matrix operand order repaired and checked by reading, not compiled, not run on hardware
//
#include <hip/hip_runtime.h>
#include <hip/hip_bf16.h>
#include <stdint.h>

#define D_MODEL 1024
#define N_HEADS 16
#define D_K     64
#define SEQ     2048
#define BATCH   4

typedef __bf16 bf16;
typedef __attribute__((ext_vector_type(16))) __bf16 v16bf;
typedef __attribute__((ext_vector_type(8)))  float  v8f;

union BFrag { v16bf v; uint4 q[2]; };

__device__ inline bf16 f2bf(float f) {
  union { float f; uint32_t u; } a; a.f = f;
  uint32_t r = a.u + 0x7FFFu + ((a.u >> 16) & 1u);  // RNE
  union { uint16_t s; bf16 b; } o; o.s = (uint16_t)(r >> 16);
  return o.b;
}

// ---- global->reg and reg->LDS staging (fp32 converts to bf16) ----
__device__ inline float4 ldg(const float* p) { return *(const float4*)p; }
__device__ inline uint2  ldg(const bf16* p)  { return *(const uint2*)p; }
__device__ inline void st_lds(float4 v, bf16* l) {
  union { bf16 b[4]; uint2 u; } t;
  t.b[0] = f2bf(v.x); t.b[1] = f2bf(v.y); t.b[2] = f2bf(v.z); t.b[3] = f2bf(v.w);
  *(uint2*)l = t.u;
}
__device__ inline void st_lds(uint2 v, bf16* l) { *(uint2*)l = v; }

// ---- pure-VALU butterfly reductions across the 16-lane C-layout rows ----
template<int CTRL>
__device__ inline float dpp_xor(float x) {   // quad_perm xor1 (0xB1) / xor2 (0x4E)
  int r = __builtin_amdgcn_update_dpp(__float_as_int(x), __float_as_int(x),
                                      CTRL, 0xF, 0xF, false);
  return __int_as_float(r);
}
__device__ inline float pl16(float x, unsigned s0, unsigned s1) {  // v_permlane16_b32
  unsigned v = __float_as_uint(x);
  return __uint_as_float(__builtin_amdgcn_permlane16(v, v, s0, s1, false, false));
}
__device__ inline float red_max16(float x) {
  x = fmaxf(x, dpp_xor<0xB1>(x));
  x = fmaxf(x, dpp_xor<0x4E>(x));
  x = fmaxf(x, pl16(x, 0x32107654u, 0xBA98FEDCu));  // xor4
  x = fmaxf(x, pl16(x, 0xFEDCBA98u, 0x76543210u));  // xor8
  return x;
}
__device__ inline float red_sum16(float x) {
  x += dpp_xor<0xB1>(x);
  x += dpp_xor<0x4E>(x);
  x += pl16(x, 0x32107654u, 0xBA98FEDCu);
  x += pl16(x, 0xFEDCBA98u, 0x76543210u);
  return x;
}

// ---- CDNA5 async copy: global -> LDS, tracked by ASYNCcnt ----
__device__ inline void async_b128(unsigned lds_off, const void* g) {
  asm volatile("global_load_async_to_lds_b128 %0, %1, off"
               :: "v"(lds_off), "v"(g) : "memory");
}
__device__ inline void wait_async0() {
  asm volatile("s_wait_asynccnt 0" ::: "memory");
}

// C = A[M,K] * W[N,K]^T, register-staged ping-pong double buffering.
// MODE 0: scatter to Q/K head buffers; MODE 1: V transposed; MODE 2: fp32 + bias.
template<int MODE, typename AT>
__global__ __launch_bounds__(256)
void gemm_bf16_wmma(const AT* __restrict__ A, const float* __restrict__ W,
                    float* __restrict__ outF, bf16* __restrict__ out0,
                    bf16* __restrict__ out1, const float* __restrict__ bias,
                    int N, int K)
{
  const int LDA = 40;                       // bf16 pitch: 16B aligned, conflict-free
  __shared__ __align__(16) bf16 sA[2][128 * 40];
  __shared__ __align__(16) bf16 sW[2][128 * 40];

  const int t    = threadIdx.x;
  const int wid  = t >> 5, lane = t & 31;
  const int wr   = wid >> 2, wc = wid & 3;  // 2x4 wave grid -> 64x32 per wave
  const int half = lane >> 4, lm = lane & 15;
  const int m0   = blockIdx.x * 128, n0 = blockIdx.y * 128;

  v8f acc[4][2] = {};

  const int rowA = t >> 3, cg = t & 7;
  const AT*    Ap = A + (size_t)(m0 + rowA) * K + 4 * cg;
  const float* Wp = W + (size_t)(n0 + rowA) * K + 4 * cg;

  using AV = decltype(ldg((const AT*)0));
  AV     ra[4];
  float4 rw[4];
  #pragma unroll
  for (int rr = 0; rr < 4; ++rr) {          // prologue: tile 0 -> regs
    ra[rr] = ldg(Ap + (size_t)(32 * rr) * K);
    rw[rr] = ldg(Wp + (size_t)(32 * rr) * K);
  }

  int buf = 0;
  for (int k0 = 0; k0 < K; k0 += 32) {
    #pragma unroll
    for (int rr = 0; rr < 4; ++rr) {
      st_lds(ra[rr], &sA[buf][(rowA + 32 * rr) * LDA + 4 * cg]);
      st_lds(rw[rr], &sW[buf][(rowA + 32 * rr) * LDA + 4 * cg]);
    }
    __syncthreads();
    if (k0 + 32 < K) {                      // overlap next tile's loads with WMMA
      #pragma unroll
      for (int rr = 0; rr < 4; ++rr) {
        ra[rr] = ldg(Ap + (size_t)(32 * rr) * K + k0 + 32);
        rw[rr] = ldg(Wp + (size_t)(32 * rr) * K + k0 + 32);
      }
    }

    BFrag b[2];
    #pragma unroll
    for (int j = 0; j < 2; ++j) {
      const bf16* p = &sW[buf][(wc * 32 + j * 16 + lm) * LDA];
      b[j].q[0] = *(const uint4*)(p + half * 8);
      b[j].q[1] = *(const uint4*)(p + 16 + half * 8);
    }
    #pragma unroll
    for (int mt = 0; mt < 4; ++mt) {
      BFrag a;
      const bf16* p = &sA[buf][(wr * 64 + mt * 16 + lm) * LDA];
      a.q[0] = *(const uint4*)(p + half * 8);
      a.q[1] = *(const uint4*)(p + 16 + half * 8);
      #pragma unroll
      for (int j = 0; j < 2; ++j)
        acc[mt][j] = __builtin_amdgcn_wmma_f32_16x16x32_bf16(
            false, a.v, false, b[j].v, (short)0, acc[mt][j], false, false);
    }
    buf ^= 1;
  }

  #pragma unroll
  for (int mt = 0; mt < 4; ++mt)
    #pragma unroll
    for (int j = 0; j < 2; ++j)
      #pragma unroll
      for (int g = 0; g < 8; ++g) {
        int m = m0 + wr * 64 + mt * 16 + g + 8 * half;
        int n = n0 + wc * 32 + j * 16 + lm;
        float v = acc[mt][j][g];
        if (MODE == 2) {
          outF[(size_t)m * N + n] = v + bias[n];
        } else if (MODE == 0) {
          int bb = m >> 11, s = m & (SEQ - 1);
          int head = n >> 7, r = n & 127;
          size_t base = ((size_t)(bb * N_HEADS + head) * SEQ + s) * D_K;
          if (r < 64) out0[base + r] = f2bf(v);
          else        out1[base + (r - 64)] = f2bf(v);
        } else {
          int bb = m >> 11, s = m & (SEQ - 1);
          int head = n >> 6, d = n & 63;
          out0[((size_t)(bb * N_HEADS + head) * D_K + d) * SEQ + s] = f2bf(v);
        }
      }
}

// Flash attention: wave owns 16 query rows; block shares 64-key K/V tiles,
// staged by async-to-LDS copies, double-buffered, one barrier per tile.
__global__ __launch_bounds__(256)
void flash_attn(const bf16* __restrict__ Q, const bf16* __restrict__ Kb,
                const bf16* __restrict__ Vt, bf16* __restrict__ Ao)
{
  const int LT = 72;                               // padded pitch (144B rows)
  __shared__ __align__(16) bf16 sK[2][64 * 72];    // 64 keys x 64 d
  __shared__ __align__(16) bf16 sV[2][64 * 72];    // 64 d    x 64 keys (pre-transposed)
  __shared__ __align__(16) bf16 sP[8 * 16 * 72];   // per-wave P scratch

  const int t    = threadIdx.x, wid = t >> 5, lane = t & 31;
  const int half = lane >> 4, lm = lane & 15;
  const int bh   = blockIdx.x;
  const int q0   = blockIdx.y * 128 + wid * 16;

  BFrag aq0, aq1;                                  // Q[16,64] as two K=32 A-frags
  {
    const bf16* qp = Q + ((size_t)bh * SEQ + q0 + lm) * D_K;
    aq0.q[0] = *(const uint4*)(qp + half * 8);
    aq0.q[1] = *(const uint4*)(qp + 16 + half * 8);
    aq1.q[0] = *(const uint4*)(qp + 32 + half * 8);
    aq1.q[1] = *(const uint4*)(qp + 48 + half * 8);
  }

  float mrow[8], lrow[8];
  #pragma unroll
  for (int g = 0; g < 8; ++g) { mrow[g] = -3.0e30f; lrow[g] = 0.0f; }
  v8f acc[4] = {};

  bf16* myP = &sP[wid * 16 * LT];
  const bf16* Kbase = Kb + (size_t)bh * SEQ * D_K;
  const bf16* Vbase = Vt + (size_t)bh * D_K * SEQ;

  // staging map: 512 x 16B chunks per tile, 2 per thread
  const int r0 = t >> 3,          c0 = t & 7;            // chunk t
  const int r1 = (t + 256) >> 3,  c1 = (t + 256) & 7;    // chunk t+256

  const int NT = SEQ / 64;
  // prologue: async-issue tile 0 into buffer 0
  {
    async_b128((unsigned)(uintptr_t)&sK[0][r0 * LT + c0 * 8],
               Kbase + (size_t)r0 * D_K + c0 * 8);
    async_b128((unsigned)(uintptr_t)&sK[0][r1 * LT + c1 * 8],
               Kbase + (size_t)r1 * D_K + c1 * 8);
    async_b128((unsigned)(uintptr_t)&sV[0][r0 * LT + c0 * 8],
               Vbase + (size_t)r0 * SEQ + c0 * 8);
    async_b128((unsigned)(uintptr_t)&sV[0][r1 * LT + c1 * 8],
               Vbase + (size_t)r1 * SEQ + c1 * 8);
  }

  int buf = 0;
  for (int kt = 0; kt < NT; ++kt) {
    wait_async0();                 // own tile-kt copies complete
    __syncthreads();               // everyone's copies visible
    if (kt + 1 < NT) {             // issue tile kt+1 into other buffer
      int nb = buf ^ 1, kk = (kt + 1) * 64;
      async_b128((unsigned)(uintptr_t)&sK[nb][r0 * LT + c0 * 8],
                 Kbase + (size_t)(kk + r0) * D_K + c0 * 8);
      async_b128((unsigned)(uintptr_t)&sK[nb][r1 * LT + c1 * 8],
                 Kbase + (size_t)(kk + r1) * D_K + c1 * 8);
      async_b128((unsigned)(uintptr_t)&sV[nb][r0 * LT + c0 * 8],
                 Vbase + (size_t)r0 * SEQ + kk + c0 * 8);
      async_b128((unsigned)(uintptr_t)&sV[nb][r1 * LT + c1 * 8],
                 Vbase + (size_t)r1 * SEQ + kk + c1 * 8);
    }

    // S = Q * K_tile^T : four 16x16 tiles (64 keys), K-dim 64 = 2 WMMAs each
    v8f s[4];
    #pragma unroll
    for (int j = 0; j < 4; ++j) {
      BFrag b0, b1;
      const bf16* kp = &sK[buf][(j * 16 + lm) * LT];
      b0.q[0] = *(const uint4*)(kp + half * 8);
      b0.q[1] = *(const uint4*)(kp + 16 + half * 8);
      b1.q[0] = *(const uint4*)(kp + 32 + half * 8);
      b1.q[1] = *(const uint4*)(kp + 48 + half * 8);
      v8f z = {};
      z    = __builtin_amdgcn_wmma_f32_16x16x32_bf16(false, aq0.v, false, b0.v, (short)0, z, false, false);
      s[j] = __builtin_amdgcn_wmma_f32_16x16x32_bf16(false, aq1.v, false, b1.v, (short)0, z, false, false);
    }

    // online softmax over 64 keys (pure-VALU row reductions)
    const float scale = 0.125f;    // 1/sqrt(64)
    #pragma unroll
    for (int g = 0; g < 8; ++g) {
      float x0 = s[0][g] * scale, x1 = s[1][g] * scale;
      float x2 = s[2][g] * scale, x3 = s[3][g] * scale;
      float mx = red_max16(fmaxf(fmaxf(x0, x1), fmaxf(x2, x3)));
      float mnew  = fmaxf(mrow[g], mx);
      float alpha = __expf(mrow[g] - mnew);
      mrow[g] = mnew;
      float p0 = __expf(x0 - mnew), p1 = __expf(x1 - mnew);
      float p2 = __expf(x2 - mnew), p3 = __expf(x3 - mnew);
      lrow[g] = lrow[g] * alpha + red_sum16((p0 + p1) + (p2 + p3));
      #pragma unroll
      for (int j = 0; j < 4; ++j) acc[j][g] *= alpha;
      int prow = g + 8 * half;     // C layout -> A layout via per-wave LDS bounce
      myP[prow * LT + lm]      = f2bf(p0);
      myP[prow * LT + 16 + lm] = f2bf(p1);
      myP[prow * LT + 32 + lm] = f2bf(p2);
      myP[prow * LT + 48 + lm] = f2bf(p3);
    }

    BFrag ap0, ap1;                // same-wave LDS is in-order: safe RAW
    {
      const bf16* pp = &myP[lm * LT];
      ap0.q[0] = *(const uint4*)(pp + half * 8);
      ap0.q[1] = *(const uint4*)(pp + 16 + half * 8);
      ap1.q[0] = *(const uint4*)(pp + 32 + half * 8);
      ap1.q[1] = *(const uint4*)(pp + 48 + half * 8);
    }
    #pragma unroll
    for (int j = 0; j < 4; ++j) {  // O += P[16,64] * V[64,64]
      BFrag bv0, bv1;
      const bf16* vp = &sV[buf][(j * 16 + lm) * LT];
      bv0.q[0] = *(const uint4*)(vp + half * 8);
      bv0.q[1] = *(const uint4*)(vp + 16 + half * 8);
      bv1.q[0] = *(const uint4*)(vp + 32 + half * 8);
      bv1.q[1] = *(const uint4*)(vp + 48 + half * 8);
      v8f tacc = __builtin_amdgcn_wmma_f32_16x16x32_bf16(false, ap0.v, false, bv0.v, (short)0, acc[j], false, false);
      acc[j]   = __builtin_amdgcn_wmma_f32_16x16x32_bf16(false, ap1.v, false, bv1.v, (short)0, tacc,   false, false);
    }
    buf ^= 1;
  }

  const int b = bh >> 4, h = bh & 15;
  #pragma unroll
  for (int j = 0; j < 4; ++j)
    #pragma unroll
    for (int g = 0; g < 8; ++g) {
      int srow = q0 + g + 8 * half;
      int col  = h * D_K + j * 16 + lm;
      float v  = acc[j][g] / lrow[g];
      Ao[(size_t)(b * SEQ + srow) * D_MODEL + col] = f2bf(v);
    }
}

extern "C" void kernel_launch(void* const* d_in, const int* in_sizes, int n_in,
                              void* d_out, int out_size, void* d_ws, size_t ws_size,
                              hipStream_t stream) {
  (void)in_sizes; (void)n_in; (void)out_size; (void)ws_size;
  const float* h_source = (const float*)d_in[0];
  const float* h_target = (const float*)d_in[1];
  const float* w_qk     = (const float*)d_in[2];
  const float* w_v      = (const float*)d_in[3];
  const float* w_o      = (const float*)d_in[4];
  const float* b_o      = (const float*)d_in[5];
  float* out = (float*)d_out;

  const size_t PH = (size_t)BATCH * N_HEADS * SEQ * D_K;   // 8,388,608 elems (16MB bf16)
  bf16* Qb  = (bf16*)d_ws;
  bf16* Kb  = Qb + PH;
  bf16* Vb  = Kb + PH;   // transposed [B,H,64,S]
  bf16* Aob = Vb + PH;   // [B,S,1024]

  dim3 blk(256);
  gemm_bf16_wmma<0, float><<<dim3(64, 16), blk, 0, stream>>>(h_source, w_qk, nullptr, Qb, Kb, nullptr, 2048, 1024);
  gemm_bf16_wmma<1, float><<<dim3(64, 8),  blk, 0, stream>>>(h_target, w_v,  nullptr, Vb, nullptr, nullptr, 1024, 1024);
  flash_attn<<<dim3(64, 16), blk, 0, stream>>>(Qb, Kb, Vb, Aob);
  gemm_bf16_wmma<2, bf16><<<dim3(64, 8),   blk, 0, stream>>>(Aob, w_o, out, nullptr, nullptr, b_o, 1024, 1024);
}
